// StructuredAttention_17437567222001
// MI455X (gfx1250) — matrix operand order repaired and checked by reading, not compiled, not run on hardware
//
#include <hip/hip_runtime.h>
#include <hip/hip_bf16.h>

#define BB   64
#define NN   512
#define SEM  256
#define STR  256

typedef __attribute__((ext_vector_type(16))) __bf16 v16bf;
typedef __attribute__((ext_vector_type(4)))  __bf16 v4bf;
typedef __attribute__((ext_vector_type(8)))  float  v8f;
typedef __attribute__((ext_vector_type(4)))  float  v4f;

// ---- WMMA fragment gathers (CDNA5 ISA 7.12.2, wave32), bf16 sources ----

// A-matrix 16x32 bf16. Element e<8 -> K = 8h+e ; e>=8 -> K = 16+8h+(e-8).
// Two contiguous 8-element (16B) runs per lane -> ds_load_b128 pairs.
__device__ __forceinline__ v16bf load_a_bf(const __bf16* src, int ld, int lane) {
    int h = (lane >> 4) & 1, m = lane & 15;
    const __bf16* p = src + m * ld + 8 * h;
    v16bf a;
#pragma unroll
    for (int i = 0; i < 8; ++i) { a[i] = p[i]; a[8 + i] = p[16 + i]; }
    return a;
}

// B-matrix 32x16 where fragment row n is contiguous along K: e<8 -> K = 8h+e ; e>=8 -> K = 16+8h+(e-8).
__device__ __forceinline__ v16bf load_b_bf(const __bf16* W, int ldw, int lane) {
    int h = (lane >> 4) & 1, n = lane & 15;
    const __bf16* p = W + n * ldw + 8 * h;
    v16bf b;
#pragma unroll
    for (int e = 0; e < 8; ++e) { b[e] = p[e]; b[8 + e] = p[16 + e]; }
    return b;
}

// ---------------------------------------------------------------------------
// Kernel W (prep): bf16-convert weights into ws (Wbil pre-transposed so K is
// the fast axis); zero T/R accumulators. 256 blocks x 256 threads.
// ---------------------------------------------------------------------------
__global__ void __launch_bounds__(256) kW(const float* __restrict__ Wp,
                                          const float* __restrict__ Wbil,
                                          const float* __restrict__ Wfz,
                                          __bf16* __restrict__ wp_bf,
                                          __bf16* __restrict__ wbilT_bf,
                                          __bf16* __restrict__ wfz_bf,
                                          float* __restrict__ T_buf,
                                          float* __restrict__ R_buf) {
    int idx = blockIdx.x * 256 + threadIdx.x;           // 0..65535
    wp_bf[idx] = (__bf16)Wp[idx];                       // row-major (C = X*Wp^T)
    int e = idx >> 8, k = idx & 255;
    wbilT_bf[idx] = (__bf16)Wbil[k * 256 + e];          // wbilT[e][k] = Wbil[k][e]
    wfz_bf[idx]         = (__bf16)Wfz[idx];
    wfz_bf[idx + 65536] = (__bf16)Wfz[idx + 65536];
    if (idx < BB * SEM) R_buf[idx] = 0.0f;
    if (idx < BB)       T_buf[idx] = 0.0f;
}

// ---------------------------------------------------------------------------
// Kernel A: per 32-row tile (2 M-tiles share each B fragment):
//   tp = str @ Wp^T + bp (WMMA) ; u = tp @ Wbil (WMMA), s = rowsum(u*tp)+bbil
//   f  = exp(str @ Wfi^T + bfi)
// ---------------------------------------------------------------------------
__global__ void __launch_bounds__(256) kA(const float* __restrict__ input,
                                          const __bf16* __restrict__ wp_bf,
                                          const float* __restrict__ bp,
                                          const __bf16* __restrict__ wbilT_bf,
                                          const float* __restrict__ bbil,
                                          const float* __restrict__ Wfi,
                                          const float* __restrict__ bfi,
                                          float* __restrict__ s_out,
                                          float* __restrict__ f_out) {
    __shared__ __bf16 str_lds[32 * STR];
    __shared__ __bf16 tp_lds[32 * STR];
    __shared__ float  s_acc[32];

    const int tid  = threadIdx.x;
    const int lane = tid & 31;
    const int wave = tid >> 5;
    const long row0 = (long)blockIdx.x * 32;
    const float* strp = input + row0 * (SEM + STR) + SEM;

    // stage 32 str rows as bf16 (float4 in, packed 4x bf16 out)
    for (int i = tid * 4; i < 32 * STR; i += 256 * 4) {
        int r = i >> 8, k = i & 255;
        v4f x = *(const v4f*)&strp[(long)r * (SEM + STR) + k];
        v4bf y;
        y[0] = (__bf16)x[0]; y[1] = (__bf16)x[1];
        y[2] = (__bf16)x[2]; y[3] = (__bf16)x[3];
        *(v4bf*)&str_lds[i] = y;
    }
    if (tid < 32) s_acc[tid] = 0.0f;
    __syncthreads();

    const int h = (lane >> 4) & 1, n = lane & 15;

    // GEMM1: tp = str @ Wp^T + bp ; 8 waves x 2 j-tiles x 2 m-tiles
#pragma unroll
    for (int t = 0; t < 2; ++t) {
        int j0 = wave * 32 + t * 16;
        v8f c0 = {}, c1 = {};
#pragma unroll 4
        for (int k0 = 0; k0 < STR; k0 += 32) {
            v16bf b  = load_b_bf(&wp_bf[j0 * STR + k0], STR, lane);
            v16bf a0 = load_a_bf(&str_lds[k0], STR, lane);
            v16bf a1 = load_a_bf(&str_lds[16 * STR + k0], STR, lane);
            c0 = __builtin_amdgcn_wmma_f32_16x16x32_bf16(false, a0, false, b, (short)0, c0, false, false);
            c1 = __builtin_amdgcn_wmma_f32_16x16x32_bf16(false, a1, false, b, (short)0, c1, false, false);
        }
        float bias = bp[j0 + n];
#pragma unroll
        for (int r = 0; r < 8; ++r) {
            tp_lds[(r + 8 * h) * STR + j0 + n]        = (__bf16)(c0[r] + bias);
            tp_lds[(16 + r + 8 * h) * STR + j0 + n]   = (__bf16)(c1[r] + bias);
        }
    }
    __syncthreads();

    // GEMM2: u = tp @ Wbil ; s partial = sum(u * tp) via shuffle tree
#pragma unroll
    for (int t = 0; t < 2; ++t) {
        int e0 = wave * 32 + t * 16;
        v8f c0 = {}, c1 = {};
#pragma unroll 4
        for (int k0 = 0; k0 < STR; k0 += 32) {
            v16bf b  = load_b_bf(&wbilT_bf[e0 * STR + k0], STR, lane);
            v16bf a0 = load_a_bf(&tp_lds[k0], STR, lane);
            v16bf a1 = load_a_bf(&tp_lds[16 * STR + k0], STR, lane);
            c0 = __builtin_amdgcn_wmma_f32_16x16x32_bf16(false, a0, false, b, (short)0, c0, false, false);
            c1 = __builtin_amdgcn_wmma_f32_16x16x32_bf16(false, a1, false, b, (short)0, c1, false, false);
        }
#pragma unroll
        for (int r = 0; r < 8; ++r) {
            int m0 = r + 8 * h;
            float p0 = c0[r] * (float)tp_lds[m0 * STR + e0 + n];
            float p1 = c1[r] * (float)tp_lds[(16 + m0) * STR + e0 + n];
            p0 += __shfl_xor(p0, 1, 16); p1 += __shfl_xor(p1, 1, 16);
            p0 += __shfl_xor(p0, 2, 16); p1 += __shfl_xor(p1, 2, 16);
            p0 += __shfl_xor(p0, 4, 16); p1 += __shfl_xor(p1, 4, 16);
            p0 += __shfl_xor(p0, 8, 16); p1 += __shfl_xor(p1, 8, 16);
            if (n == 0) {
                atomicAdd(&s_acc[m0], p0);
                atomicAdd(&s_acc[16 + m0], p1);
            }
        }
    }
    __syncthreads();

    if (wave == 0) {   // 32 lanes -> 32 rows
        float acc = 0.0f;
        for (int k = 0; k < STR; ++k) acc += (float)str_lds[lane * STR + k] * Wfi[k];
        f_out[row0 + lane] = __expf(acc + bfi[0]);
        s_out[row0 + lane] = s_acc[lane] + bbil[0];
    }
}

// ---------------------------------------------------------------------------
// Kernel B: per-batch reductions, 4 blocks per batch (atomic combine):
//   T[b] += sum f ;  R[b,e] += sum_{i>=1} exp(s[b,i]) * sem_v[b,i,e]
// ---------------------------------------------------------------------------
__global__ void __launch_bounds__(256) kB(const float* __restrict__ input,
                                          const float* __restrict__ s_in,
                                          const float* __restrict__ f_in,
                                          float* __restrict__ T_out,
                                          float* __restrict__ R_out) {
    const int b  = blockIdx.x >> 2;
    const int c4 = blockIdx.x & 3;
    const int e  = threadIdx.x;
    const int nlo = c4 * 128, nhi = nlo + 128;
    const float* sem = input + (long)b * NN * (SEM + STR);

    float r = 0.0f;
    for (int nn = nlo; nn < nhi; ++nn) {
        if (nn == 0) continue;
        r += __expf(s_in[b * NN + nn]) * sem[(long)nn * (SEM + STR) + e];
    }
    atomicAdd(&R_out[b * SEM + e], r);

    __shared__ float tf[128];
    if (e < 128) tf[e] = f_in[b * NN + nlo + e];
    __syncthreads();
    for (int off = 64; off > 0; off >>= 1) {
        if (e < off) tf[e] += tf[e + off];
        __syncthreads();
    }
    if (e == 0) atomicAdd(&T_out[b], tf[0]);
}

// ---------------------------------------------------------------------------
// Kernel C: closed-form pinp, finp = [sem | pinp] (32x512 bf16 LDS),
//           out = relu(finp @ Wfz^T + bfz)
// ---------------------------------------------------------------------------
__global__ void __launch_bounds__(256) kC(const float* __restrict__ input,
                                          const float* __restrict__ s_in,
                                          const float* __restrict__ f_in,
                                          const float* __restrict__ T_in,
                                          const float* __restrict__ R_in,
                                          const float* __restrict__ exparam,
                                          const __bf16* __restrict__ wfz_bf,
                                          const float* __restrict__ bfz,
                                          float* __restrict__ out) {
    __shared__ __bf16 finp[32 * 512];

    const int tid = threadIdx.x;
    const int b   = blockIdx.x >> 4;          // 16 blocks per batch
    const int n0  = (blockIdx.x & 15) * 32;
    const float* base = input + ((long)b * NN + n0) * (SEM + STR);
    const float* sem0 = input + (long)b * NN * (SEM + STR);  // row n=0 of batch

    const float T    = T_in[b];
    const float invT = 1.0f / T;
    const float f0   = f_in[b * NN];
    const float ea0  = __expf(s_in[b * NN]);
    const float invN = 1.0f / (float)NN;

    // left half: sem rows (bf16, vectorized)
    for (int i = tid * 4; i < 32 * SEM; i += 256 * 4) {
        int r = i >> 8, e = i & 255;
        v4f x = *(const v4f*)&base[(long)r * (SEM + STR) + e];
        v4bf y;
        y[0] = (__bf16)x[0]; y[1] = (__bf16)x[1];
        y[2] = (__bf16)x[2]; y[3] = (__bf16)x[3];
        *(v4bf*)&finp[r * 512 + e] = y;
    }
    // right half: pinp closed form
    for (int i = tid; i < 32 * SEM; i += 256) {
        int r = i >> 8, e = i & 255;
        int nrow = n0 + r;
        float fn = f_in[b * NN + nrow];
        float p;
        if (nrow == 0) {
            p = fn * invT * exparam[e] - invT * R_in[b * SEM + e];
        } else {
            float sn = s_in[b * NN + nrow];
            float dg = __expf(-sn) * invN * (1.0f + (f0 - fn) * invT);
            p = fn * invT * exparam[e] + ea0 * dg * sem0[e];
        }
        finp[r * 512 + SEM + e] = (__bf16)p;
    }
    __syncthreads();

    const int lane = tid & 31, wave = tid >> 5;
    const int h = (lane >> 4) & 1, n = lane & 15;
    float* outp = out + ((long)b * NN + n0) * SEM;
#pragma unroll
    for (int t = 0; t < 2; ++t) {
        int j0 = wave * 32 + t * 16;
        v8f c0 = {}, c1 = {};
#pragma unroll 4
        for (int k0 = 0; k0 < 512; k0 += 32) {
            v16bf bfrag = load_b_bf(&wfz_bf[j0 * 512 + k0], 512, lane);
            v16bf a0 = load_a_bf(&finp[k0], 512, lane);
            v16bf a1 = load_a_bf(&finp[16 * 512 + k0], 512, lane);
            c0 = __builtin_amdgcn_wmma_f32_16x16x32_bf16(false, a0, false, bfrag, (short)0, c0, false, false);
            c1 = __builtin_amdgcn_wmma_f32_16x16x32_bf16(false, a1, false, bfrag, (short)0, c1, false, false);
        }
        float bias = bfz[j0 + n];
#pragma unroll
        for (int r = 0; r < 8; ++r) {
            float v0 = c0[r] + bias;
            float v1 = c1[r] + bias;
            outp[(long)(r + 8 * h) * SEM + j0 + n]      = v0 > 0.0f ? v0 : 0.0f;
            outp[(long)(16 + r + 8 * h) * SEM + j0 + n] = v1 > 0.0f ? v1 : 0.0f;
        }
    }
}

// ---------------------------------------------------------------------------
extern "C" void kernel_launch(void* const* d_in, const int* in_sizes, int n_in,
                              void* d_out, int out_size, void* d_ws, size_t ws_size,
                              hipStream_t stream) {
    const float* input   = (const float*)d_in[0];
    const float* Wp      = (const float*)d_in[1];
    const float* bp      = (const float*)d_in[2];
    const float* Wbil    = (const float*)d_in[3];
    const float* bbil    = (const float*)d_in[4];
    const float* Wfi     = (const float*)d_in[5];
    const float* bfi     = (const float*)d_in[6];
    const float* exparam = (const float*)d_in[7];
    const float* Wfz     = (const float*)d_in[8];
    const float* bfz     = (const float*)d_in[9];

    float* ws    = (float*)d_ws;
    float* s_buf = ws;                         // B*N         (32768)
    float* f_buf = ws + BB * NN;               // B*N         (32768)
    float* T_buf = ws + 2 * BB * NN;           // B           (64)
    float* R_buf = ws + 2 * BB * NN + BB;      // B*SEM       (16384)
    __bf16* bfbase   = (__bf16*)(ws + 2 * BB * NN + BB + BB * SEM);
    __bf16* wp_bf    = bfbase;                 // 65536
    __bf16* wbilT_bf = bfbase + 65536;         // 65536
    __bf16* wfz_bf   = bfbase + 131072;        // 131072

    const int rowTiles = (BB * NN) / 32;       // 1024

    kW<<<256, 256, 0, stream>>>(Wp, Wbil, Wfz, wp_bf, wbilT_bf, wfz_bf, T_buf, R_buf);
    kA<<<rowTiles, 256, 0, stream>>>(input, wp_bf, bp, wbilT_bf, bbil, Wfi, bfi, s_buf, f_buf);
    kB<<<BB * 4, 256, 0, stream>>>(input, s_buf, f_buf, T_buf, R_buf);
    kC<<<rowTiles, 256, 0, stream>>>(input, s_buf, f_buf, T_buf, R_buf, exparam, wfz_bf, bfz, (float*)d_out);
}
